// SceneRFHead_80711025426499
// MI455X (gfx1250) — matrix operand order repaired and checked by reading, not compiled, not run on hardware
//
#include <hip/hip_runtime.h>
#include <hip/hip_bf16.h>

typedef _Float16 v16h __attribute__((ext_vector_type(16)));
typedef float    v8f  __attribute__((ext_vector_type(8)));

#define R_RAYS   4096
#define NSAMP    64
#define NFRAGS   304   // 16 (Win 64x128) + 3 blocks * 3 mats * 32 (128x128)

__device__ __forceinline__ float sigmoidf_(float x){ return 1.0f/(1.0f+expf(-x)); }

__device__ __forceinline__ v8f wmma_f16(v16h a, v16h b, v8f c){
  return __builtin_amdgcn_wmma_f32_16x16x32_f16(false, a, false, b, (short)0, c, false, false);
}

// A-fragment (16x32 f16): lane&15 = row; lanes<16 hold K {0..7,16..23}, lanes>=16 hold K {8..15,24..31}
__device__ __forceinline__ v16h load_a(const _Float16* s, int stride, int row, int khalf, int kt){
  const _Float16* p = s + row*stride + kt*32 + khalf;
  v16h a;
#pragma unroll
  for (int i=0;i<8;++i){ a[i] = p[i]; a[i+8] = p[i+16]; }
  return a;
}

// B-fragment pre-swizzled: frag f, lane l holds 16 contiguous f16
__device__ __forceinline__ v16h load_b(const _Float16* bt, int f, int lane){
  const _Float16* p = bt + (size_t)f*512 + lane*16;
  v16h b;
#pragma unroll
  for (int i=0;i<16;++i) b[i] = p[i];
  return b;
}

// C/D layout store with ReLU: VGPR r -> row mbase + r + (lane<16?0:8), col = nt*16 + (lane&15)
__device__ __forceinline__ void store_relu(_Float16* dst, const v8f* t, int mbase, int lane){
  int colb = lane & 15;
  int rofs = (lane >> 4) << 3;
#pragma unroll
  for (int nt=0; nt<8; ++nt){
#pragma unroll
    for (int r=0; r<8; ++r){
      int row = mbase + rofs + r;
      dst[row*136 + nt*16 + colb] = (_Float16)fmaxf(t[nt][r], 0.0f);
    }
  }
}

// ---------------- weight convert: f32 -> f16, pre-swizzled WMMA B fragments ----------------
__global__ void k_conv(const float* __restrict__ Win, const float* __restrict__ Wz,
                       const float* __restrict__ W0,  const float* __restrict__ W1,
                       _Float16* __restrict__ dst)
{
  int t = blockIdx.x*blockDim.x + threadIdx.x;
  if (t >= NFRAGS*32) return;
  int f = t >> 5, l = t & 31;
  int nl = l & 15, kh = (l >> 4) * 8;
  int kt, nt, kdim;
  const float* W;
  if (f < 16){ kt = f >> 3; nt = f & 7; W = Win; kdim = 42; }
  else {
    int f2 = f - 16;
    int il = f2 / 96; int rem = f2 - il*96;
    int mt = rem >> 5; int ff = rem & 31;
    kt = ff >> 3; nt = ff & 7;
    W = (mt==0 ? Wz : (mt==1 ? W0 : W1)) + il*128*128;
    kdim = 128;
  }
  int k0 = kt*32 + kh, n = nt*16 + nl;
#pragma unroll
  for (int i=0;i<16;++i){
    int k = k0 + (i < 8 ? i : i + 8);
    float v = (k < kdim) ? W[k*128 + n] : 0.0f;
    dst[(size_t)f*512 + l*16 + i] = (_Float16)v;
  }
}

__global__ void k_init(float* acc){ int t = threadIdx.x; if (t < 8) acc[t] = 0.0f; }

// ---------------- ray setup ----------------
__global__ void k_ray_setup(const float* __restrict__ camK, const float* __restrict__ s2i,
                            const float* __restrict__ pix,
                            float* __restrict__ camd, float* __restrict__ dirs,
                            float* __restrict__ vdir, float* __restrict__ ptsg,
                            float* __restrict__ vdg)
{
  int r = blockIdx.x*blockDim.x + threadIdx.x;
  if (r >= R_RAYS) return;
  float a=camK[0],b=camK[1],c=camK[2],d=camK[3],e=camK[4],f=camK[5],g=camK[6],h=camK[7],i=camK[8];
  float det = a*(e*i-f*h) - b*(d*i-f*g) + c*(d*h-e*g);
  float id = 1.0f/det;
  float inv[9];
  inv[0]=(e*i-f*h)*id; inv[1]=(c*h-b*i)*id; inv[2]=(b*f-c*e)*id;
  inv[3]=(f*g-d*i)*id; inv[4]=(a*i-c*g)*id; inv[5]=(c*d-a*f)*id;
  inv[6]=(d*h-e*g)*id; inv[7]=(b*g-a*h)*id; inv[8]=(a*e-b*d)*id;
  float px = pix[r*2+0], py = pix[r*2+1];
  float cd0 = inv[0]*px + inv[1]*py + inv[2];
  float cd1 = inv[3]*px + inv[4]*py + inv[5];
  float cd2 = inv[6]*px + inv[7]*py + inv[8];
  camd[r*3+0]=cd0; camd[r*3+1]=cd1; camd[r*3+2]=cd2;
  float dl0 = s2i[0]*cd0 + s2i[1]*cd1 + s2i[2]*cd2;
  float dl1 = s2i[4]*cd0 + s2i[5]*cd1 + s2i[6]*cd2;
  float dl2 = s2i[8]*cd0 + s2i[9]*cd1 + s2i[10]*cd2;
  dirs[r*3+0]=dl0; dirs[r*3+1]=dl1; dirs[r*3+2]=dl2;
  float vn = 1.0f/sqrtf(dl0*dl0+dl1*dl1+dl2*dl2);
  float v0=dl0*vn, v1=dl1*vn, v2=dl2*vn;
  vdir[r*3+0]=v0; vdir[r*3+1]=v1; vdir[r*3+2]=v2;
  float o0=s2i[3], o1=s2i[7], o2=s2i[11];
#pragma unroll
  for (int gq=0; gq<4; ++gq){
    float dq = ((float)gq + 0.5f)*25.0f;
    int idx = (r*4+gq)*3;
    ptsg[idx+0]=o0+dl0*dq; ptsg[idx+1]=o1+dl1*dq; ptsg[idx+2]=o2+dl2*dq;
    vdg[idx+0]=v0; vdg[idx+1]=v1; vdg[idx+2]=v2;
  }
}

// ---------------- fused WMMA decode: trilinear z + posenc x -> 3-block ResNet MLP ----------------
template<int OUT>
__global__ __launch_bounds__(128) void k_decode(
    const float* __restrict__ pts, const float* __restrict__ vd,
    const float* __restrict__ bev, const _Float16* __restrict__ Bt,
    const float* __restrict__ bin, const float* __restrict__ bz,
    const float* __restrict__ b0,  const float* __restrict__ b1,
    const float* __restrict__ Wout,const float* __restrict__ bout,
    float* __restrict__ out)
{
  __shared__ _Float16 sZ[64*136];
  __shared__ _Float16 sX[64*72];
  __shared__ _Float16 sA0[64*136];
  __shared__ _Float16 sA1[64*136];
  const int tid = threadIdx.x;

  // ---- stage A: 2 threads per point; compute trilinear z (f16) and posenc x (f16)
  {
    const int m = tid >> 1, half = tid & 1;
    const int gp = blockIdx.x*64 + m;
    const float p0 = pts[gp*3+0], p1 = pts[gp*3+1], p2 = pts[gp*3+2];
    float gx = (p0 + 51.2f)*(199.0f/102.4f);
    float gy = (p1 + 51.2f)*(199.0f/102.4f);
    float gz = (p2 + 5.0f)*(7.0f/8.0f);
    float valid = (gx>=0.f && gx<=199.f && gy>=0.f && gy<=199.f && gz>=0.f && gz<=7.f) ? 1.0f : 0.0f;
    int x0 = (int)fminf(fmaxf(floorf(gx),0.f),198.f);
    int y0 = (int)fminf(fmaxf(floorf(gy),0.f),198.f);
    int z0 = (int)fminf(fmaxf(floorf(gz),0.f),6.f);
    float fx = fminf(fmaxf(gx-(float)x0,0.f),1.f);
    float fy = fminf(fmaxf(gy-(float)y0,0.f),1.f);
    float fz = fminf(fmaxf(gz-(float)z0,0.f),1.f);
    float ox=1.f-fx, oy=1.f-fy, oz=1.f-fz;
    float wc[8];
    wc[0]=oz*ox*oy; wc[1]=oz*ox*fy; wc[2]=oz*fx*oy; wc[3]=oz*fx*fy;
    wc[4]=fz*ox*oy; wc[5]=fz*ox*fy; wc[6]=fz*fx*oy; wc[7]=fz*fx*fy;
    int base[8];
#pragma unroll
    for (int k=0;k<8;++k){
      int dz=(k>>2)&1, dx=(k>>1)&1, dy=k&1;
      base[k] = (z0+dz)*40000 + (x0+dx)*200 + (y0+dy);
    }
    // vox[a,h,w,c] = bev[(c*8+a)*40000 + h*200 + w]; this thread does channels [half*64, half*64+64)
    for (int c=0;c<64;++c){
      int cc = half*64 + c;
      const float* bp = bev + (size_t)cc*320000;
      float s = 0.0f;
#pragma unroll
      for (int k=0;k<8;++k) s += wc[k]*bp[base[k]];
      sZ[m*136 + cc] = (_Float16)(s*valid);
    }
    if (half == 0){
      sX[m*72+0]=(_Float16)p0; sX[m*72+1]=(_Float16)p1; sX[m*72+2]=(_Float16)p2;
#pragma unroll
      for (int fq=0; fq<6; ++fq){
        float fr = (float)(1<<fq)*3.14159265358979f;
        float pv[3] = {p0,p1,p2};
#pragma unroll
        for (int dd=0; dd<3; ++dd){
          sX[m*72 + 3 + fq*6 + dd]     = (_Float16)sinf(pv[dd]*fr);
          sX[m*72 + 3 + fq*6 + 3 + dd] = (_Float16)cosf(pv[dd]*fr);
        }
      }
      sX[m*72+39]=(_Float16)vd[gp*3+0];
      sX[m*72+40]=(_Float16)vd[gp*3+1];
      sX[m*72+41]=(_Float16)vd[gp*3+2];
      for (int k=42;k<64;++k) sX[m*72+k]=(_Float16)0.0f;
    }
  }
  __syncthreads();

  // ---- WMMA stage: wave w handles rows [w*16, w*16+16), all 128 cols
  const int lane  = tid & 31;
  const int mbase = (tid >> 5) * 16;
  const int row   = mbase + (lane & 15);
  const int khalf = (lane >> 4) * 8;
  const int colb  = lane & 15;

  v16h aX0 = load_a(sX, 72, row, khalf, 0);
  v16h aX1 = load_a(sX, 72, row, khalf, 1);
  v16h aZ[4];
#pragma unroll
  for (int kt=0; kt<4; ++kt) aZ[kt] = load_a(sZ, 136, row, khalf, kt);

  v8f h[8];
#pragma unroll
  for (int nt=0; nt<8; ++nt){
    v8f acc = {0.f,0.f,0.f,0.f,0.f,0.f,0.f,0.f};
    acc = wmma_f16(aX0, load_b(Bt, 0*8+nt, lane), acc);
    acc = wmma_f16(aX1, load_b(Bt, 1*8+nt, lane), acc);
    float bv = bin[nt*16 + colb];
#pragma unroll
    for (int r=0;r<8;++r) acc[r] += bv;
    h[nt] = acc;
  }

  for (int i=0; i<3; ++i){
    const int fb = 16 + i*96;
    // h += z @ Wz[i] + bz[i]
#pragma unroll
    for (int nt=0; nt<8; ++nt){
      v8f acc = h[nt];
#pragma unroll
      for (int kt=0; kt<4; ++kt) acc = wmma_f16(aZ[kt], load_b(Bt, fb + kt*8 + nt, lane), acc);
      float bv = bz[i*128 + nt*16 + colb];
#pragma unroll
      for (int r=0;r<8;++r) acc[r] += bv;
      h[nt] = acc;
    }
    store_relu(sA0, h, mbase, lane);
    __syncthreads();
    v16h aH[4];
#pragma unroll
    for (int kt=0; kt<4; ++kt) aH[kt] = load_a(sA0, 136, row, khalf, kt);
    // net = relu(h) @ W0[i] + b0[i]
    v8f nb[8];
#pragma unroll
    for (int nt=0; nt<8; ++nt){
      v8f acc = {0.f,0.f,0.f,0.f,0.f,0.f,0.f,0.f};
#pragma unroll
      for (int kt=0; kt<4; ++kt) acc = wmma_f16(aH[kt], load_b(Bt, fb + 32 + kt*8 + nt, lane), acc);
      float bv = b0[i*128 + nt*16 + colb];
#pragma unroll
      for (int r=0;r<8;++r) acc[r] += bv;
      nb[nt] = acc;
    }
    store_relu(sA1, nb, mbase, lane);
    __syncthreads();
    v16h aN[4];
#pragma unroll
    for (int kt=0; kt<4; ++kt) aN[kt] = load_a(sA1, 136, row, khalf, kt);
    // h += relu(net) @ W1[i] + b1[i]
#pragma unroll
    for (int nt=0; nt<8; ++nt){
      v8f acc = {0.f,0.f,0.f,0.f,0.f,0.f,0.f,0.f};
#pragma unroll
      for (int kt=0; kt<4; ++kt) acc = wmma_f16(aN[kt], load_b(Bt, fb + 64 + kt*8 + nt, lane), acc);
      float bv = b1[i*128 + nt*16 + colb];
      v8f hh = h[nt];
#pragma unroll
      for (int r=0;r<8;++r) hh[r] += acc[r] + bv;
      h[nt] = hh;
    }
    __syncthreads();
  }

  // out = relu(h) @ Wout + bout (tiny N -> scalar from LDS)
  store_relu(sA0, h, mbase, lane);
  __syncthreads();
  for (int idx = tid; idx < 64*OUT; idx += 128){
    int mm = idx / OUT, o = idx - mm*OUT;
    float s = bout[o];
    for (int k=0;k<128;++k) s += (float)sA0[mm*136 + k] * Wout[k*OUT + o];
    out[(size_t)(blockIdx.x*64 + mm)*OUT + o] = s;
  }
}

// ---------------- gaussian-head postprocess + sample build + sort ----------------
__global__ void k_build_samples(const float* __restrict__ outg, const float* __restrict__ s2i,
                                const float* __restrict__ dirs, const float* __restrict__ vdir,
                                float* __restrict__ gm, float* __restrict__ gs,
                                float* __restrict__ dall, float* __restrict__ ptsm,
                                float* __restrict__ vdm)
{
  int r = blockIdx.x*blockDim.x + threadIdx.x;
  if (r >= R_RAYS) return;
  float gmv[4], gsv[4];
#pragma unroll
  for (int g=0; g<4; ++g){
    float o0 = outg[(r*4+g)*2+0], o1 = outg[(r*4+g)*2+1];
    gmv[g] = ((float)g + 0.5f)*25.0f + tanhf(o0)*12.5f;
    gsv[g] = sigmoidf_(o1)*5.0f + 0.001f;
    gm[r*4+g]=gmv[g]; gs[r*4+g]=gsv[g];
  }
  float d[NSAMP];
  for (int i=0;i<32;++i) d[i] = 0.5f + (float)i*(99.5f/31.0f);
  for (int g=0; g<4; ++g)
    for (int j=0;j<8;++j)
      d[32+g*8+j] = gmv[g] + (-1.0f + (float)j*(2.0f/7.0f))*2.0f*gsv[g];
  for (int i=0;i<NSAMP;++i) d[i] = fminf(fmaxf(d[i],0.01f),100.0f);
  for (int a=1;a<NSAMP;++a){
    float key = d[a]; int b = a-1;
    while (b>=0 && d[b]>key){ d[b+1]=d[b]; --b; }
    d[b+1]=key;
  }
  float o0=s2i[3], o1=s2i[7], o2=s2i[11];
  float dl0=dirs[r*3+0], dl1=dirs[r*3+1], dl2=dirs[r*3+2];
  float v0=vdir[r*3+0], v1=vdir[r*3+1], v2=vdir[r*3+2];
  for (int s=0;s<NSAMP;++s){
    dall[r*NSAMP+s] = d[s];
    int idx = (r*NSAMP+s)*3;
    ptsm[idx+0]=o0+dl0*d[s]; ptsm[idx+1]=o1+dl1*d[s]; ptsm[idx+2]=o2+dl2*d[s];
    vdm[idx+0]=v0; vdm[idx+1]=v1; vdm[idx+2]=v2;
  }
}

__device__ __forceinline__ void sample3(const float* __restrict__ img, float u, float v, float* out3){
  float x0c = fminf(fmaxf(floorf(u),0.f),1598.f);
  float y0c = fminf(fmaxf(floorf(v),0.f),898.f);
  int x0=(int)x0c, y0=(int)y0c;
  float fx = fminf(fmaxf(u-x0c,0.f),1.f);
  float fy = fminf(fmaxf(v-y0c,0.f),1.f);
#pragma unroll
  for (int c=0;c<3;++c){
    const float* b = img + (size_t)c*1440000 + (size_t)y0*1600 + x0;
    float v00=b[0], v01=b[1], v10=b[1600], v11=b[1601];
    out3[c] = v00*(1.f-fx)*(1.f-fy) + v01*fx*(1.f-fy) + v10*(1.f-fx)*fy + v11*fx*fy;
  }
}

// ---------------- volume rendering + losses ----------------
__global__ void k_render(const float* __restrict__ outm, const float* __restrict__ dall,
                         const float* __restrict__ gm, const float* __restrict__ gs,
                         const float* __restrict__ camd, const float* __restrict__ srcimg,
                         const float* __restrict__ tgtimg, const float* __restrict__ camK,
                         const float* __restrict__ s2t, const float* __restrict__ pix,
                         float* __restrict__ acc, float* __restrict__ out)
{
  int r = blockIdx.x*blockDim.x + threadIdx.x;
  if (r >= R_RAYS) return;
  float d[NSAMP], w[NSAMP], q[NSAMP];
  for (int s=0;s<NSAMP;++s) d[s] = dall[r*NSAMP+s];
  float T=1.0f, sw=0.0f, dep=0.0f, c0=0.0f, c1=0.0f, c2=0.0f;
  for (int s=0;s<NSAMP;++s){
    const float* o = outm + (size_t)(r*NSAMP+s)*4;
    float sg = fmaxf(o[0],0.0f);
    float delta = (s<NSAMP-1) ? (d[s+1]-d[s]) : 1e10f;
    float alpha = 1.0f - expf(-sg*delta);
    float wv = alpha*T;
    w[s]=wv; sw+=wv;
    dep += wv*d[s];
    c0 += wv*sigmoidf_(o[1]); c1 += wv*sigmoidf_(o[2]); c2 += wv*sigmoidf_(o[3]);
    T *= (1.0f - alpha + 1e-10f);
  }
  float qsum=0.0f;
  for (int s=0;s<NSAMP;++s){
    float qv=0.0f;
#pragma unroll
    for (int g=0; g<4; ++g){
      float t = (d[s]-gm[r*4+g])/gs[r*4+g];
      qv += expf(-0.5f*t*t - logf(gs[r*4+g]));
    }
    qv *= 0.25f; q[s]=qv; qsum+=qv;
  }
  float kl=0.0f;
  for (int s=0;s<NSAMP;++s){
    float pw = w[s]/(sw+1e-6f);
    float qn = q[s]/(qsum+1e-6f);
    kl += pw*(logf(pw+1e-6f) - logf(qn+1e-6f));
  }
  float d2c=1e30f;
#pragma unroll
  for (int g=0; g<4; ++g) d2c = fminf(d2c, fabsf(gm[r*4+g]-dep));
  float src[3]; sample3(srcimg, pix[r*2+0], pix[r*2+1], src);
  float lc = fabsf(src[0]-c0)+fabsf(src[1]-c1)+fabsf(src[2]-c2);
  // reprojection
  float cp0=camd[r*3+0]*dep, cp1=camd[r*3+1]*dep, cp2=camd[r*3+2]*dep;
  float t0 = s2t[0]*cp0 + s2t[1]*cp1 + s2t[2]*cp2 + s2t[3];
  float t1 = s2t[4]*cp0 + s2t[5]*cp1 + s2t[6]*cp2 + s2t[7];
  float t2 = s2t[8]*cp0 + s2t[9]*cp1 + s2t[10]*cp2 + s2t[11];
  float p0 = camK[0]*t0 + camK[1]*t1 + camK[2]*t2;
  float p1 = camK[3]*t0 + camK[4]*t1 + camK[5]*t2;
  float p2 = camK[6]*t0 + camK[7]*t1 + camK[8]*t2;
  float u = p0/(p2+1e-6f), v = p1/(p2+1e-6f);
  float tc[3]; sample3(tgtimg, u, v, tc);
  float dm = (fabsf(tc[0]-src[0])+fabsf(tc[1]-src[1])+fabsf(tc[2]-src[2]))*(1.0f/3.0f);
  float mask = (dep < 30.0f) ? 1.0f : 0.0f;
  atomicAdd(&acc[0], kl);
  atomicAdd(&acc[1], d2c);
  atomicAdd(&acc[2], lc);
  atomicAdd(&acc[3], dm*mask);
  atomicAdd(&acc[4], mask);
  out[1+r] = dep;
  out[1+R_RAYS + r*3+0]=c0; out[1+R_RAYS + r*3+1]=c1; out[1+R_RAYS + r*3+2]=c2;
}

__global__ void k_finalize(const float* __restrict__ acc, float* __restrict__ out){
  if (threadIdx.x==0)
    out[0] = acc[0]*(1.0f/R_RAYS) + 0.01f*acc[1]*(1.0f/R_RAYS)
           + acc[2]*(1.0f/(R_RAYS*3.0f)) + acc[3]/(acc[4]+1e-6f);
}

extern "C" void kernel_launch(void* const* d_in, const int* in_sizes, int n_in,
                              void* d_out, int out_size, void* d_ws, size_t ws_size,
                              hipStream_t stream)
{
  const float* bev  = (const float*)d_in[0];
  const float* src  = (const float*)d_in[1];
  const float* tgt  = (const float*)d_in[2];
  const float* camK = (const float*)d_in[3];
  const float* s2i  = (const float*)d_in[4];
  const float* s2t  = (const float*)d_in[5];
  const float* pix  = (const float*)d_in[6];
  const float* m1_Win=(const float*)d_in[7];  const float* m1_bin=(const float*)d_in[8];
  const float* m1_Wz =(const float*)d_in[9];  const float* m1_bz =(const float*)d_in[10];
  const float* m1_W0 =(const float*)d_in[11]; const float* m1_b0 =(const float*)d_in[12];
  const float* m1_W1 =(const float*)d_in[13]; const float* m1_b1 =(const float*)d_in[14];
  const float* m1_Wout=(const float*)d_in[15];const float* m1_bout=(const float*)d_in[16];
  const float* m2_Win=(const float*)d_in[17]; const float* m2_bin=(const float*)d_in[18];
  const float* m2_Wz =(const float*)d_in[19]; const float* m2_bz =(const float*)d_in[20];
  const float* m2_W0 =(const float*)d_in[21]; const float* m2_b0 =(const float*)d_in[22];
  const float* m2_W1 =(const float*)d_in[23]; const float* m2_b1 =(const float*)d_in[24];
  const float* m2_Wout=(const float*)d_in[25];const float* m2_bout=(const float*)d_in[26];

  float* ws = (float*)d_ws;
  size_t off = 0;
  float* acc  = ws + off; off += 8;
  float* camd = ws + off; off += (size_t)R_RAYS*3;
  float* dirs = ws + off; off += (size_t)R_RAYS*3;
  float* vdir = ws + off; off += (size_t)R_RAYS*3;
  float* ptsg = ws + off; off += (size_t)R_RAYS*4*3;
  float* vdg  = ws + off; off += (size_t)R_RAYS*4*3;
  float* outg = ws + off; off += (size_t)R_RAYS*4*2;
  float* gm   = ws + off; off += (size_t)R_RAYS*4;
  float* gs   = ws + off; off += (size_t)R_RAYS*4;
  float* dall = ws + off; off += (size_t)R_RAYS*NSAMP;
  float* ptsm = ws + off; off += (size_t)R_RAYS*NSAMP*3;
  float* vdm  = ws + off; off += (size_t)R_RAYS*NSAMP*3;
  float* outm = ws + off; off += (size_t)R_RAYS*NSAMP*4;
  _Float16* bt1 = (_Float16*)(ws + off);
  _Float16* bt2 = bt1 + (size_t)NFRAGS*512;

  k_conv<<<(NFRAGS*32+255)/256, 256, 0, stream>>>(m1_Win, m1_Wz, m1_W0, m1_W1, bt1);
  k_conv<<<(NFRAGS*32+255)/256, 256, 0, stream>>>(m2_Win, m2_Wz, m2_W0, m2_W1, bt2);
  k_init<<<1, 32, 0, stream>>>(acc);
  k_ray_setup<<<R_RAYS/256, 256, 0, stream>>>(camK, s2i, pix, camd, dirs, vdir, ptsg, vdg);
  k_decode<2><<<R_RAYS*4/64, 128, 0, stream>>>(ptsg, vdg, bev, bt2,
      m2_bin, m2_bz, m2_b0, m2_b1, m2_Wout, m2_bout, outg);
  k_build_samples<<<R_RAYS/128, 128, 0, stream>>>(outg, s2i, dirs, vdir, gm, gs, dall, ptsm, vdm);
  k_decode<4><<<R_RAYS*NSAMP/64, 128, 0, stream>>>(ptsm, vdm, bev, bt1,
      m1_bin, m1_bz, m1_b0, m1_b1, m1_Wout, m1_bout, outm);
  k_render<<<R_RAYS/128, 128, 0, stream>>>(outm, dall, gm, gs, camd, src, tgt,
      camK, s2t, pix, acc, (float*)d_out);
  k_finalize<<<1, 1, 0, stream>>>(acc, (float*)d_out);
}
